// DownBlock_48816598286943
// MI455X (gfx1250) — matrix operand order repaired and checked by reading, not compiled, not run on hardware
//
#include <hip/hip_runtime.h>

// ---------------------------------------------------------------------------
// Types for WMMA (gfx1250, wave32)
// ---------------------------------------------------------------------------
typedef __attribute__((ext_vector_type(16))) __bf16    v16bf;
typedef __attribute__((ext_vector_type(8)))  float     v8f;
typedef __attribute__((ext_vector_type(4)))  unsigned  v4u;

union Frag32B { v16bf bf; v4u u[2]; };   // 32-byte WMMA A/B fragment per lane

__device__ __forceinline__ unsigned short f2bf(float f) {
  unsigned u = __float_as_uint(f);
  u = (u + 0x7FFFu + ((u >> 16) & 1u)) >> 16;   // round-to-nearest-even
  return (unsigned short)u;
}

// ---------------------------------------------------------------------------
// Problem constants (match reference)
// ---------------------------------------------------------------------------
#define BB    4
#define PP    4096
#define NN    (BB * PP)        // 16384
#define NDN   (NN / 8)         // 2048
#define KT27  27
#define CIN   128
#define COUT  256
#define GG    32
#define HH    16
#define DHD   16
#define LCTXN 77
#define CTXN  768

// ---------------------------------------------------------------------------
// Weight convert+transpose: in [KT, Cin, Cout] f32 -> out [KT, Cout, Cin] bf16
// ---------------------------------------------------------------------------
__global__ void __launch_bounds__(256)
wconv_kernel(const float* __restrict__ in, unsigned short* __restrict__ out,
             int KT, int Cin, int Cout) {
  long total = (long)KT * Cin * Cout;
  long i = (long)blockIdx.x * 256 + threadIdx.x;
  if (i >= total) return;
  long per = (long)Cin * Cout;
  int  k  = (int)(i / per);
  long r  = i % per;
  int  co = (int)(r / Cin);
  int  ci = (int)(r % Cin);
  out[i] = f2bf(in[(long)k * per + (long)ci * Cout + co]);
}

// ---------------------------------------------------------------------------
// GroupNorm + SiLU -> bf16 (optionally also emit bf16 copy of raw input)
// grid = B*G blocks, 256 threads; per-(sample,group) stats over P x cpg
// ---------------------------------------------------------------------------
__global__ void __launch_bounds__(256)
gn_silu_kernel(const float* __restrict__ x, const float* __restrict__ gamma,
               const float* __restrict__ beta, unsigned short* __restrict__ out,
               unsigned short* __restrict__ xcopy, int C, int G, int P) {
  __shared__ float s_sum[256], s_sq[256];
  const int tid = threadIdx.x;
  const int b = blockIdx.x / G, g = blockIdx.x % G;
  const int cpg = C / G;
  const long base = (long)b * P * C;
  const int cnt = P * cpg;
  float sum = 0.f, sq = 0.f;
  for (int i = tid; i < cnt; i += 256) {
    int r = i / cpg, ch = g * cpg + (i % cpg);
    long o = base + (long)r * C + ch;
    float v = x[o];
    sum += v; sq += v * v;
    if (xcopy) xcopy[o] = f2bf(v);
  }
  s_sum[tid] = sum; s_sq[tid] = sq; __syncthreads();
  for (int s = 128; s > 0; s >>= 1) {
    if (tid < s) { s_sum[tid] += s_sum[tid + s]; s_sq[tid] += s_sq[tid + s]; }
    __syncthreads();
  }
  const float mean = s_sum[0] / (float)cnt;
  const float var  = s_sq[0] / (float)cnt - mean * mean;
  const float inv  = rsqrtf(var + 1e-5f);
  for (int i = tid; i < cnt; i += 256) {
    int r = i / cpg, ch = g * cpg + (i % cpg);
    long o = base + (long)r * C + ch;
    float v = (x[o] - mean) * inv * gamma[ch] + beta[ch];
    out[o] = f2bf(v / (1.f + __expf(-v)));   // silu
  }
}

// ---------------------------------------------------------------------------
// Row LayerNorm -> bf16. grid = N rows, block = C = 256 threads.
// ---------------------------------------------------------------------------
__global__ void __launch_bounds__(256)
ln_kernel(const float* __restrict__ x, const float* __restrict__ w,
          const float* __restrict__ b, unsigned short* __restrict__ out, int C) {
  __shared__ float s_sum[256], s_sq[256];
  const int tid = threadIdx.x;
  const long row = blockIdx.x;
  float v = x[row * C + tid];
  s_sum[tid] = v; s_sq[tid] = v * v; __syncthreads();
  for (int s = 128; s > 0; s >>= 1) {
    if (tid < s) { s_sum[tid] += s_sum[tid + s]; s_sq[tid] += s_sq[tid + s]; }
    __syncthreads();
  }
  const float mean = s_sum[0] / (float)C;
  const float var  = s_sq[0] / (float)C - mean * mean;
  const float xn   = (v - mean) * rsqrtf(var + 1e-5f) * w[tid] + b[tid];
  out[row * C + tid] = f2bf(xn);
}

// ---------------------------------------------------------------------------
// Context k/v projection: ctx [B*LCTX, CTX] f32 @ wk/wv [CTX, C] + bias -> f32
// grid = B*LCTX, 256 threads; context row staged in LDS; weight reads coalesced
// ---------------------------------------------------------------------------
__global__ void __launch_bounds__(256)
ctxproj_kernel(const float* __restrict__ ctx,
               const float* __restrict__ wk, const float* __restrict__ bk,
               const float* __restrict__ wv, const float* __restrict__ bv,
               float* __restrict__ kout, float* __restrict__ vout,
               int CTXd, int C) {
  __shared__ float sc[CTXN];
  const int bl = blockIdx.x;
  const float* crow = ctx + (long)bl * CTXd;
  for (int i = threadIdx.x; i < CTXd; i += 256) sc[i] = crow[i];
  __syncthreads();
  const int d = threadIdx.x;
  float ak = 0.f, av = 0.f;
  for (int e = 0; e < CTXd; ++e) {
    float c = sc[e];
    ak = fmaf(c, wk[(long)e * C + d], ak);
    av = fmaf(c, wv[(long)e * C + d], av);
  }
  kout[(long)bl * C + d] = ak + bk[d];
  vout[(long)bl * C + d] = av + bv[d];
}

// ---------------------------------------------------------------------------
// Cross-attention core: online softmax over 77 context tokens, per (point, head)
// grid = (N/128, H), block = 128; K/V head-slice staged in LDS; output bf16
// ---------------------------------------------------------------------------
__global__ void __launch_bounds__(128)
attn_kernel(const float* __restrict__ q, const float* __restrict__ kctx,
            const float* __restrict__ vctx, unsigned short* __restrict__ obf,
            int P, int Lc, int C, float scale) {
  __shared__ float sk[LCTXN * DHD], sv[LCTXN * DHD];
  const int h = blockIdx.y;
  const int p = blockIdx.x * 128 + threadIdx.x;
  const int b = p / P;                       // 128 | P -> uniform per block
  for (int i = threadIdx.x; i < Lc * DHD; i += 128) {
    int l = i / DHD, d = i % DHD;
    long src = ((long)(b * Lc + l)) * C + h * DHD + d;
    sk[i] = kctx[src];
    sv[i] = vctx[src];
  }
  __syncthreads();
  float qv[DHD];
#pragma unroll
  for (int d = 0; d < DHD; ++d) qv[d] = q[(long)p * C + h * DHD + d];
  float m = -1e30f, den = 0.f, o[DHD];
#pragma unroll
  for (int d = 0; d < DHD; ++d) o[d] = 0.f;
  for (int l = 0; l < Lc; ++l) {
    float s = 0.f;
#pragma unroll
    for (int d = 0; d < DHD; ++d) s = fmaf(qv[d], sk[l * DHD + d], s);
    s *= scale;
    float nm = fmaxf(m, s);
    float corr = __expf(m - nm);
    float w = __expf(s - nm);
    den = den * corr + w;
#pragma unroll
    for (int d = 0; d < DHD; ++d) o[d] = o[d] * corr + w * sv[l * DHD + d];
    m = nm;
  }
  const float r = 1.f / den;
#pragma unroll
  for (int d = 0; d < DHD; ++d) obf[(long)p * C + h * DHD + d] = f2bf(o[d] * r);
}

// ---------------------------------------------------------------------------
// Generic gathered-GEMM via WMMA bf16:
//   out[m, :] = sum_k sum_c act[neigh[m,k], c] * wt[k, :, c]  (+bias, +residual)
// GATHER=false -> identity gather (dense GEMM rows).
// Block: 128 threads (4 waves); block tile = 32 rows x 256 cols; each wave
// owns 32 rows x 64 cols = 8 accumulators, so every B fragment is reused by
// two A fragments (6 fragment loads per 8 WMMAs). A/B fragments are loaded
// straight from global in the ISA VGPR layout (weights pre-transposed to
// [K][Cout][Cin] bf16).
// ---------------------------------------------------------------------------
template <int CIN_T, bool GATHER>
__global__ void __launch_bounds__(128)
conv_wmma_kernel(const int* __restrict__ neigh, int ktaps,
                 const unsigned short* __restrict__ act,
                 const unsigned short* __restrict__ wt,
                 const float* __restrict__ bias,
                 const float* __restrict__ residual,
                 float* __restrict__ out_f32,
                 unsigned short* __restrict__ out_bf16,
                 int Cout) {
  const int lane = threadIdx.x & 31;
  const int wid  = threadIdx.x >> 5;
  const int m0   = blockIdx.x * 32;           // 32 rows per block
  const int hi   = lane >> 4;                 // half-wave selector
  const int ln15 = lane & 15;
  const int colBase = wid * 64;

  v8f acc[2][4];
#pragma unroll
  for (int rt = 0; rt < 2; ++rt)
#pragma unroll
    for (int nt = 0; nt < 4; ++nt) acc[rt][nt] = v8f{};

  const v4u vz = {0u, 0u, 0u, 0u};

  for (int k = 0; k < ktaps; ++k) {
    long r0, r1;
    bool ok0 = true, ok1 = true;
    if (GATHER) {
      int i0 = neigh[(long)(m0 + ln15) * ktaps + k];
      int i1 = neigh[(long)(m0 + 16 + ln15) * ktaps + k];
      ok0 = (i0 >= 0); ok1 = (i1 >= 0);
      r0 = ok0 ? i0 : 0;
      r1 = ok1 ? i1 : 0;
    } else {
      r0 = m0 + ln15;
      r1 = r0 + 16;
    }
    // A rows: lanes 0-15 carry K 0-7 / 16-23, lanes 16-31 carry K 8-15 / 24-31
    const unsigned short* a0  = act + r0 * CIN_T + hi * 8;
    const unsigned short* a1  = act + r1 * CIN_T + hi * 8;
    const unsigned short* wk_ = wt + (long)k * Cout * CIN_T + hi * 8;

#pragma unroll
    for (int c = 0; c < CIN_T; c += 32) {
      Frag32B A0, A1;
      if (ok0) { A0.u[0] = *(const v4u*)(a0 + c); A0.u[1] = *(const v4u*)(a0 + c + 16); }
      else     { A0.u[0] = vz; A0.u[1] = vz; }
      if (ok1) { A1.u[0] = *(const v4u*)(a1 + c); A1.u[1] = *(const v4u*)(a1 + c + 16); }
      else     { A1.u[0] = vz; A1.u[1] = vz; }
#pragma unroll
      for (int nt = 0; nt < 4; ++nt) {
        // B fragment: lane = col N; half-wave carries K 8h..8h+7 / 16+8h..16+8h+7
        const unsigned short* bp = wk_ + (long)(colBase + nt * 16 + ln15) * CIN_T + c;
        Frag32B Bf;
        Bf.u[0] = *(const v4u*)(bp);
        Bf.u[1] = *(const v4u*)(bp + 16);
        acc[0][nt] = __builtin_amdgcn_wmma_f32_16x16x32_bf16(
            false, A0.bf, false, Bf.bf, (short)0, acc[0][nt], false, false);
        acc[1][nt] = __builtin_amdgcn_wmma_f32_16x16x32_bf16(
            false, A1.bf, false, Bf.bf, (short)0, acc[1][nt], false, false);
      }
    }
  }

  // Epilogue: C/D layout — VGPR r -> row r + 8*hi, col = lane&15.
#pragma unroll
  for (int rt = 0; rt < 2; ++rt) {
    const int rbase = m0 + rt * 16 + hi * 8;
#pragma unroll
    for (int nt = 0; nt < 4; ++nt) {
      const int col = colBase + nt * 16 + ln15;
      const float bv = bias ? bias[col] : 0.f;
#pragma unroll
      for (int r = 0; r < 8; ++r) {
        long o = (long)(rbase + r) * Cout + col;
        float v = acc[rt][nt][r] + bv;
        if (residual) v += residual[o];
        if (out_f32)  out_f32[o] = v;
        if (out_bf16) out_bf16[o] = f2bf(v);
      }
    }
  }
}

// ---------------------------------------------------------------------------
// Launcher
// ---------------------------------------------------------------------------
extern "C" void kernel_launch(void* const* d_in, const int* in_sizes, int n_in,
                              void* d_out, int out_size, void* d_ws, size_t ws_size,
                              hipStream_t stream) {
  (void)in_sizes; (void)n_in; (void)out_size; (void)ws_size;

  const float* x        = (const float*)d_in[0];
  const float* context  = (const float*)d_in[1];
  const int*   neigh    = (const int*)d_in[2];
  const int*   neigh_dn = (const int*)d_in[3];
  const float* g1_w = (const float*)d_in[4];
  const float* g1_b = (const float*)d_in[5];
  const float* conv1_w = (const float*)d_in[6];
  const float* conv1_b = (const float*)d_in[7];
  const float* g2_w = (const float*)d_in[8];
  const float* g2_b = (const float*)d_in[9];
  const float* conv2_w = (const float*)d_in[10];
  const float* conv2_b = (const float*)d_in[11];
  const float* sc_w = (const float*)d_in[12];
  const float* sc_b = (const float*)d_in[13];
  const float* ln_w = (const float*)d_in[14];
  const float* ln_b = (const float*)d_in[15];
  const float* wq = (const float*)d_in[16];
  const float* bq = (const float*)d_in[17];
  const float* wk = (const float*)d_in[18];
  const float* bk = (const float*)d_in[19];
  const float* wv = (const float*)d_in[20];
  const float* bv = (const float*)d_in[21];
  const float* wo = (const float*)d_in[22];
  const float* bo = (const float*)d_in[23];
  const float* down_w = (const float*)d_in[24];
  const float* down_b = (const float*)d_in[25];

  float* out_xdown = (float*)d_out;                         // [ND, COUT]
  float* out_h     = (float*)d_out + (long)NDN * COUT;      // [N, COUT]

  // ---- workspace bump allocator (deterministic layout each call) ----
  char* cur = (char*)d_ws;
  auto alloc = [&](size_t bytes) -> char* {
    char* r = cur;
    cur += (bytes + 255) & ~(size_t)255;
    return r;
  };
  typedef unsigned short u16;
  u16* Wt1   = (u16*)alloc((size_t)KT27 * CIN  * COUT * 2);
  u16* Wt2   = (u16*)alloc((size_t)KT27 * COUT * COUT * 2);
  u16* WtD   = (u16*)alloc((size_t)KT27 * COUT * COUT * 2);
  u16* scT   = (u16*)alloc((size_t)CIN  * COUT * 2);
  u16* wqT   = (u16*)alloc((size_t)COUT * COUT * 2);
  u16* woT   = (u16*)alloc((size_t)COUT * COUT * 2);
  u16* h1bf  = (u16*)alloc((size_t)NN * CIN  * 2);   // silu(gn1(x))  bf16
  u16* xbf   = (u16*)alloc((size_t)NN * CIN  * 2);   // x             bf16
  u16* h2bf  = (u16*)alloc((size_t)NN * COUT * 2);   // silu(gn2(h1)) bf16
  u16* xnbf  = (u16*)alloc((size_t)NN * COUT * 2);   // ln(h_pre)     bf16
  u16* obf   = (u16*)alloc((size_t)NN * COUT * 2);   // attn out      bf16
  u16* hfbf  = (u16*)alloc((size_t)NN * COUT * 2);   // h_final       bf16
  float* hc1   = (float*)alloc((size_t)NN * COUT * 4); // conv1 out
  float* hsc   = (float*)alloc((size_t)NN * COUT * 4); // shortcut
  float* h_pre = (float*)alloc((size_t)NN * COUT * 4); // resblock out
  float* qbuf  = (float*)alloc((size_t)NN * COUT * 4); // q proj
  float* kctx  = (float*)alloc((size_t)BB * LCTXN * COUT * 4);
  float* vctx  = (float*)alloc((size_t)BB * LCTXN * COUT * 4);

  auto launch_wconv = [&](const float* in, u16* out, int KT, int Ci, int Co) {
    long total = (long)KT * Ci * Co;
    wconv_kernel<<<dim3((unsigned)((total + 255) / 256)), 256, 0, stream>>>(
        in, out, KT, Ci, Co);
  };

  // 1) weight transposes (f32 -> bf16 [K][Cout][Cin])
  launch_wconv(conv1_w, Wt1, KT27, CIN,  COUT);
  launch_wconv(conv2_w, Wt2, KT27, COUT, COUT);
  launch_wconv(down_w,  WtD, KT27, COUT, COUT);
  launch_wconv(sc_w,    scT, 1,    CIN,  COUT);
  launch_wconv(wq,      wqT, 1,    COUT, COUT);
  launch_wconv(wo,      woT, 1,    COUT, COUT);

  // 2) GN1 + silu -> h1bf; also emit bf16 copy of x for shortcut GEMM
  gn_silu_kernel<<<dim3(BB * GG), 256, 0, stream>>>(x, g1_w, g1_b, h1bf, xbf,
                                                    CIN, GG, PP);
  // 3) conv1 (gathered WMMA GEMM) -> hc1
  conv_wmma_kernel<CIN, true><<<dim3(NN / 32), 128, 0, stream>>>(
      neigh, KT27, h1bf, Wt1, conv1_b, nullptr, hc1, nullptr, COUT);
  // 4) GN2 + silu -> h2bf
  gn_silu_kernel<<<dim3(BB * GG), 256, 0, stream>>>(hc1, g2_w, g2_b, h2bf,
                                                    nullptr, COUT, GG, PP);
  // 5) shortcut: x @ sc_w + sc_b -> hsc
  conv_wmma_kernel<CIN, false><<<dim3(NN / 32), 128, 0, stream>>>(
      nullptr, 1, xbf, scT, sc_b, nullptr, hsc, nullptr, COUT);
  // 6) conv2 + shortcut residual -> h_pre
  conv_wmma_kernel<COUT, true><<<dim3(NN / 32), 128, 0, stream>>>(
      neigh, KT27, h2bf, Wt2, conv2_b, hsc, h_pre, nullptr, COUT);
  // 7) LayerNorm -> xnbf
  ln_kernel<<<dim3(NN), 256, 0, stream>>>(h_pre, ln_w, ln_b, xnbf, COUT);
  // 8) q projection (WMMA) -> qbuf
  conv_wmma_kernel<COUT, false><<<dim3(NN / 32), 128, 0, stream>>>(
      nullptr, 1, xnbf, wqT, bq, nullptr, qbuf, nullptr, COUT);
  // 9) context k/v projections
  ctxproj_kernel<<<dim3(BB * LCTXN), 256, 0, stream>>>(
      context, wk, bk, wv, bv, kctx, vctx, CTXN, COUT);
  // 10) attention core (online softmax) -> obf
  attn_kernel<<<dim3(NN / 128, HH), 128, 0, stream>>>(
      qbuf, kctx, vctx, obf, PP, LCTXN, COUT, 0.25f);
  // 11) out projection + residual(h_pre) -> out_h (f32) and hfbf (bf16)
  conv_wmma_kernel<COUT, false><<<dim3(NN / 32), 128, 0, stream>>>(
      nullptr, 1, obf, woT, bo, h_pre, out_h, hfbf, COUT);
  // 12) stride-2 downsample conv -> out_xdown
  conv_wmma_kernel<COUT, true><<<dim3(NDN / 32), 128, 0, stream>>>(
      neigh_dn, KT27, hfbf, WtD, down_b, nullptr, out_xdown, nullptr, COUT);
}
